// Gemma2Attention_20590073217358
// MI455X (gfx1250) — matrix operand order repaired and checked by reading, not compiled, not run on hardware
//
#include <hip/hip_runtime.h>

// ---------------------------------------------------------------------------
// Gemma2 attention block for gfx1250 (MI455X), bf16 WMMA + async LDS staging.
// ---------------------------------------------------------------------------

typedef unsigned short u16;
typedef __attribute__((ext_vector_type(16))) __bf16  v16bf;
typedef __attribute__((ext_vector_type(8)))  float   v8f;
typedef __attribute__((ext_vector_type(4)))  unsigned int u32x4;

// Problem constants (from reference)
constexpr int BB   = 2;
constexpr int S    = 2048;
constexpr int HID  = 2304;
constexpr int H    = 8;
constexpr int KVH  = 4;
constexpr int D    = 256;
constexpr int HD   = H * D;     // 2048
constexpr int WIN  = 1024;
constexpr float SCALING = 0.0625f;          // 256^-0.5
constexpr float SOFTCAP = 50.0f;

// ---------------------------------------------------------------------------
// Helpers
// ---------------------------------------------------------------------------
__device__ __forceinline__ u16 f2bf(float f) {
    unsigned int u = __float_as_uint(f);
    u += 0x7FFFu + ((u >> 16) & 1u);        // round-to-nearest-even
    return (u16)(u >> 16);
}
__device__ __forceinline__ float bf2f(u16 h) {
    return __uint_as_float(((unsigned int)h) << 16);
}

// A-operand tile fragment: 8 contiguous bf16 at p, 8 more at p+16
__device__ __forceinline__ v16bf load_a_frag(const u16* p) {
    union { v16bf v; u32x4 u[2]; } t;
    t.u[0] = *reinterpret_cast<const u32x4*>(p);
    t.u[1] = *reinterpret_cast<const u32x4*>(p + 16);
    return t.v;
}
// B-operand tile fragment: 8 contiguous bf16 at p, 8 more at p+16
__device__ __forceinline__ v16bf load_b_frag(const u16* p) {
    union { v16bf v; u32x4 u[2]; } t;
    t.u[0] = *reinterpret_cast<const u32x4*>(p);
    t.u[1] = *reinterpret_cast<const u32x4*>(p + 16);
    return t.v;
}
__device__ __forceinline__ v8f wmma_bf16(v16bf a, v16bf b, v8f c) {
    return __builtin_amdgcn_wmma_f32_16x16x32_bf16(false, a, false, b,
                                                   (short)0, c, false, false);
}

// LDS byte offset of a generic pointer to __shared__ (flat addr low 32 bits
// are the LDS segment offset on gfx1250's aperture scheme).
__device__ __forceinline__ unsigned lds_off(const void* p) {
    return (unsigned)(unsigned long long)p;
}

// Async global -> LDS copy, 16 bytes per lane (VDST = LDS byte address VGPR,
// address pair = global address). Tracked by ASYNCcnt.
__device__ __forceinline__ void async_ld_b128(unsigned lds_addr, const u16* g) {
    asm volatile("global_load_async_to_lds_b128 %0, %1, off"
                 :: "v"(lds_addr), "v"(g) : "memory");
}
__device__ __forceinline__ void wait_async_le8() {
    asm volatile("s_wait_asynccnt 0x8" ::: "memory");
}
__device__ __forceinline__ void wait_async_0() {
    asm volatile("s_wait_asynccnt 0x0" ::: "memory");
}

// branch-free softcap: SOFTCAP * tanh(x/SOFTCAP); tanh(y) = 1 - 2/(e^{2y}+1)
__device__ __forceinline__ float softcap(float x) {
    float t = __expf(x * (2.0f / SOFTCAP));
    return SOFTCAP * (1.0f - 2.0f / (t + 1.0f));
}

// ---------------------------------------------------------------------------
// fp32 -> bf16 conversion
// ---------------------------------------------------------------------------
__global__ void f32_to_bf16_kernel(const float* __restrict__ in,
                                   u16* __restrict__ out, int n) {
    int i = blockIdx.x * blockDim.x + threadIdx.x;
    int stride = gridDim.x * blockDim.x;
    for (; i < n; i += stride) out[i] = f2bf(in[i]);
}

// ---------------------------------------------------------------------------
// Tiled bf16 WMMA GEMM: C[M,N] = A[M,K] * W[N,K]^T
//   MODE 0: bf16 row-major  |  MODE 1: bf16 V-transposed  |  MODE 2: fp32
// Block: 256 threads = 8 waves, block tile 64(M) x 128(N), wave tile 16x64.
// ---------------------------------------------------------------------------
template <int MODE>
__global__ __launch_bounds__(256) void gemm_bf16_kernel(
    const u16* __restrict__ A, const u16* __restrict__ Wm,
    void* __restrict__ Cout, int M, int N, int K) {
    const int lane = threadIdx.x & 31;
    const int w    = threadIdx.x >> 5;
    const int r    = lane & 15;
    const int half = lane >> 4;
    const int ak   = half * 8;       // A operand k-offset
    const int bk   = half * 8;       // B operand k-offset

    const int m0 = blockIdx.y * 64  + (w & 3) * 16;
    const int n0 = blockIdx.x * 128 + (w >> 2) * 64;

    const u16* Arow = A + (size_t)(m0 + r) * K;

    v8f acc[4];
    const v8f vzero = {0, 0, 0, 0, 0, 0, 0, 0};
#pragma unroll
    for (int j = 0; j < 4; j++) acc[j] = vzero;

    for (int kb = 0; kb < K; kb += 32) {
        v16bf at = load_a_frag(Arow + kb + ak);
        if (kb + 256 < K)
            __builtin_prefetch(Arow + kb + 256, 0, 1);
#pragma unroll
        for (int j = 0; j < 4; j++) {
            const u16* wp = Wm + (size_t)(n0 + j * 16 + r) * K + kb + bk;
            acc[j] = wmma_bf16(at, load_b_frag(wp), acc[j]);
        }
    }

#pragma unroll
    for (int j = 0; j < 4; j++) {
#pragma unroll
        for (int v = 0; v < 8; v++) {
            int m = m0 + v + half * 8;
            int n = n0 + j * 16 + r;
            float val = acc[j][v];
            if (MODE == 0) {
                ((u16*)Cout)[(size_t)m * N + n] = f2bf(val);
            } else if (MODE == 1) {
                int b   = m >> 11;          // S = 2048
                int s   = m & 2047;
                int kvh = n >> 8;           // D = 256
                int d   = n & 255;
                ((u16*)Cout)[((size_t)((b * KVH + kvh) * D + d) << 11) + s] = f2bf(val);
            } else {
                ((float*)Cout)[(size_t)m * N + n] = val;
            }
        }
    }
}

// ---------------------------------------------------------------------------
// RoPE: in [B*S, NH*D] bf16 -> out [B, NH, S, D] bf16
// ---------------------------------------------------------------------------
__global__ __launch_bounds__(128) void rope_kernel(
    const u16* __restrict__ in, u16* __restrict__ out,
    const int* __restrict__ pos_ids, int NH) {
    const int s = blockIdx.x, hh = blockIdx.y, b = blockIdx.z;
    const int d = threadIdx.x;

    const u16* ip = in + ((size_t)(b * S + s) * NH + hh) * D;
    float x1 = bf2f(ip[d]);
    float x2 = bf2f(ip[d + 128]);

    float p = (float)pos_ids[s];
    float ang = p * __expf(-(float)d * (1.0f / 128.0f) * 9.210340371976184f);
    float c = cosf(ang), sn = sinf(ang);

    u16* op = out + ((size_t)(b * NH + hh) * S + s) * D;
    op[d]       = f2bf(x1 * c - x2 * sn);
    op[d + 128] = f2bf(x2 * c + x1 * sn);
}

// ---------------------------------------------------------------------------
// Flash attention, sliding window + tanh softcap.
// Grid: (S/64, H, B); block 128 (= 4 waves). One wave = one 16-row Q tile.
// All 4 waves sweep the block-union key window so K/V chunks are staged into
// LDS ONCE per block via global_load_async_to_lds_b128 (ASYNCcnt), K double-
// buffered, V awaited only after QK+softmax. 2 barriers per 32-key chunk.
// ---------------------------------------------------------------------------
__global__ __launch_bounds__(128) void attn_kernel(
    const u16* __restrict__ Qr,   // [B,H,S,D]   bf16 (post-RoPE)
    const u16* __restrict__ Kr,   // [B,KVH,S,D] bf16 (post-RoPE)
    const u16* __restrict__ Vt,   // [B,KVH,D,S] bf16
    u16* __restrict__ attn_out) { // [B*S, H*D]  bf16
    constexpr int KSTR = D + 8;   // 264 elems (528B rows, 16B aligned, depadded banks)
    constexpr int VSTR = 40;      // 80B rows
    constexpr int PSTR = 40;      // 80B rows
    __shared__ __align__(16) u16 kbuf[2][32 * KSTR];  // 33,792 B
    __shared__ __align__(16) u16 vbuf[D * VSTR];      // 20,480 B
    __shared__ __align__(16) u16 plds[4][16 * PSTR];  //  5,120 B

    const int tid  = threadIdx.x;
    const int lane = tid & 31;
    const int w    = tid >> 5;
    const int r    = lane & 15;
    const int half = lane >> 4;
    const int ak   = half * 8;     // A-operand k-offset
    const int bk   = half * 8;     // B-operand k-offset

    const int h   = blockIdx.y, b = blockIdx.z;
    const int q0b = blockIdx.x * 64;       // block's first query row
    const int q0  = q0b + w * 16;          // this wave's query tile
    const int kvh = h >> 1;                // n_rep = H/KVH = 2

    const u16* Qb = Qr + ((size_t)(b * H + h) * S + q0) * D;
    const u16* Kb = Kr + (size_t)(b * KVH + kvh) * S * D;
    const u16* Vb = Vt + (size_t)(b * KVH + kvh) * D * S;

    const unsigned koff0 = lds_off(&kbuf[0][0]);
    const unsigned koff1 = lds_off(&kbuf[1][0]);
    const unsigned voff  = lds_off(&vbuf[0]);

    v8f o[16];
    const v8f vzero = {0, 0, 0, 0, 0, 0, 0, 0};
#pragma unroll
    for (int i = 0; i < 16; i++) o[i] = vzero;
    float mi[8], li[8];
#pragma unroll
    for (int i = 0; i < 8; i++) { mi[i] = -3.0e38f; li[i] = 0.0f; }

    // Block-union chunk range (all waves iterate identically)
    const int kstart  = (q0b >= WIN - 1) ? ((q0b - (WIN - 1)) & ~31) : 0;
    const int khi_blk = q0b + 63;
    const int nchunks = (khi_blk - kstart) / 32 + 1;

    // Prologue: stage K chunk 0 into kbuf[0] (8 x b128 per thread = 16KB)
#pragma unroll
    for (int t = 0; t < 8; t++) {
        int e = t * 128 + tid;               // 1024 transfers: 32 rows x 32 blk
        int row = e >> 5, cb = e & 31;
        async_ld_b128(koff0 + (unsigned)(row * KSTR + cb * 8) * 2,
                      Kb + (size_t)(kstart + row) * D + cb * 8);
    }
    wait_async_0();
    __syncthreads();

    int cur = 0;
    for (int ci = 0; ci < nchunks; ci++) {
        const int  kb       = kstart + ci * 32;
        const bool has_next = (ci + 1 < nchunks);

        // -- issue V_cur (8 x b128/thread), then K_next (8 x b128/thread) --
#pragma unroll
        for (int t = 0; t < 8; t++) {
            int e = t * 128 + tid;           // 1024 transfers: 256 rows x 4 blk
            int row = e >> 2, cb = e & 3;
            async_ld_b128(voff + (unsigned)(row * VSTR + cb * 8) * 2,
                          Vb + (size_t)row * S + kb + cb * 8);
        }
        if (has_next) {
            const int kbn = kb + 32;
            const unsigned kn = (cur ? koff0 : koff1);
#pragma unroll
            for (int t = 0; t < 8; t++) {
                int e = t * 128 + tid;
                int row = e >> 5, cb = e & 31;
                async_ld_b128(kn + (unsigned)(row * KSTR + cb * 8) * 2,
                              Kb + (size_t)(kbn + row) * D + cb * 8);
            }
        }

        // Does this wave's window intersect the chunk? (wave-uniform)
        const bool active = (kb <= q0 + 15) && (kb + 31 >= q0 - (WIN - 1));
        float scale[8];

        if (active) {
            // ---- S = Q K^T over D=256 from resident kbuf[cur] -----------
            const u16* kc = kbuf[cur];
            v8f s0 = vzero, s1 = vzero;
#pragma unroll
            for (int c = 0; c < 8; c++) {
                v16bf qa = load_a_frag(Qb + (size_t)r * D + c * 32 + ak);
                v16bf k0 = load_b_frag(kc + (r)      * KSTR + c * 32 + bk);
                v16bf k1 = load_b_frag(kc + (16 + r) * KSTR + c * 32 + bk);
                s0 = wmma_bf16(qa, k0, s0);
                s1 = wmma_bf16(qa, k1, s1);
            }

            // ---- softcap + window mask + online softmax -----------------
#pragma unroll
            for (int v = 0; v < 8; v++) {
                int qrow = q0 + v + half * 8;
                int k0i  = kb + r, k1i = k0i + 16;
                float a0 = softcap(s0[v] * SCALING);
                float a1 = softcap(s1[v] * SCALING);
                a0 = ((k0i <= qrow) && (qrow - k0i < WIN)) ? a0 : -3.0e38f;
                a1 = ((k1i <= qrow) && (qrow - k1i < WIN)) ? a1 : -3.0e38f;

                float mx = fmaxf(a0, a1);
#pragma unroll
                for (int dm = 1; dm < 16; dm <<= 1)
                    mx = fmaxf(mx, __shfl_xor(mx, dm, 32));
                float mnew = fmaxf(mi[v], mx);
                float sc   = __expf(mi[v] - mnew);
                float p0   = __expf(a0 - mnew);
                float p1   = __expf(a1 - mnew);
                float rs   = p0 + p1;
#pragma unroll
                for (int dm = 1; dm < 16; dm <<= 1)
                    rs += __shfl_xor(rs, dm, 32);
                li[v] = li[v] * sc + rs;
                mi[v] = mnew;
                scale[v] = sc;

                int prow = v + half * 8;              // C-layout row
                plds[w][prow * PSTR + r]      = f2bf(p0);
                plds[w][prow * PSTR + 16 + r] = f2bf(p1);
            }
        }

        // ---- wait for V_cur (K_next may still be in flight: in-order) ----
        if (has_next) wait_async_le8(); else wait_async_0();
        __syncthreads();

        if (active) {
            // ---- O = diag(scale) * O + P * V from resident vbuf ----------
            v16bf pA = load_a_frag(&plds[w][r * PSTR + ak]);
#pragma unroll
            for (int nd = 0; nd < 16; nd++) {
                v16bf vfrag = load_b_frag(vbuf + (nd * 16 + r) * VSTR + bk);
#pragma unroll
                for (int v = 0; v < 8; v++) o[nd][v] *= scale[v];
                o[nd] = wmma_bf16(pA, vfrag, o[nd]);
            }
        }

        // K_next resident + everyone done reading vbuf/kbuf[cur]
        wait_async_0();
        __syncthreads();
        cur ^= 1;
    }

    // ---- normalize and store --------------------------------------------
#pragma unroll
    for (int nd = 0; nd < 16; nd++) {
#pragma unroll
        for (int v = 0; v < 8; v++) {
            int m = q0 + v + half * 8;
            float val = o[nd][v] / li[v];
            attn_out[(size_t)(b * S + m) * HD + h * D + nd * 16 + r] = f2bf(val);
        }
    }
}

// ---------------------------------------------------------------------------
// Launch
// ---------------------------------------------------------------------------
extern "C" void kernel_launch(void* const* d_in, const int* in_sizes, int n_in,
                              void* d_out, int out_size, void* d_ws, size_t ws_size,
                              hipStream_t stream) {
    (void)in_sizes; (void)n_in; (void)out_size; (void)ws_size;
    const float* X   = (const float*)d_in[0];
    const float* Wq  = (const float*)d_in[1];
    const float* Wk  = (const float*)d_in[2];
    const float* Wv  = (const float*)d_in[3];
    const float* Wo  = (const float*)d_in[4];
    const int*   pos = (const int*)d_in[5];

    char* ws = (char*)d_ws;
    size_t off = 0;
    auto alloc = [&](size_t bytes) -> void* {
        void* p = ws + off;
        off += (bytes + 255) & ~(size_t)255;
        return p;
    };
    const size_t TOK = (size_t)BB * S;               // 4096 tokens
    u16* Xbf  = (u16*)alloc(TOK * HID * 2);
    u16* Wqb  = (u16*)alloc((size_t)HD * HID * 2);
    u16* Wkb  = (u16*)alloc((size_t)KVH * D * HID * 2);
    u16* Wvb  = (u16*)alloc((size_t)KVH * D * HID * 2);
    u16* Wob  = (u16*)alloc((size_t)HID * HD * 2);
    u16* qraw = (u16*)alloc(TOK * HD * 2);
    u16* kraw = (u16*)alloc(TOK * KVH * D * 2);
    u16* VtB  = (u16*)alloc((size_t)BB * KVH * D * S * 2);
    u16* QrB  = (u16*)alloc((size_t)BB * H * S * D * 2);
    u16* KrB  = (u16*)alloc((size_t)BB * KVH * S * D * 2);
    u16* attnB= (u16*)alloc(TOK * HD * 2);

    // 1) fp32 -> bf16 conversions
    f32_to_bf16_kernel<<<2048, 256, 0, stream>>>(X,  Xbf, (int)(TOK * HID));
    f32_to_bf16_kernel<<<2048, 256, 0, stream>>>(Wq, Wqb, HD * HID);
    f32_to_bf16_kernel<<<1024, 256, 0, stream>>>(Wk, Wkb, KVH * D * HID);
    f32_to_bf16_kernel<<<1024, 256, 0, stream>>>(Wv, Wvb, KVH * D * HID);
    f32_to_bf16_kernel<<<2048, 256, 0, stream>>>(Wo, Wob, HID * HD);

    // 2) QKV projections (M=4096, K=2304)
    gemm_bf16_kernel<0><<<dim3(HD / 128,  (int)TOK / 64), 256, 0, stream>>>(
        Xbf, Wqb, qraw, (int)TOK, HD, HID);
    gemm_bf16_kernel<0><<<dim3((KVH * D) / 128, (int)TOK / 64), 256, 0, stream>>>(
        Xbf, Wkb, kraw, (int)TOK, KVH * D, HID);
    gemm_bf16_kernel<1><<<dim3((KVH * D) / 128, (int)TOK / 64), 256, 0, stream>>>(
        Xbf, Wvb, VtB, (int)TOK, KVH * D, HID);

    // 3) RoPE
    rope_kernel<<<dim3(S, H,   BB), 128, 0, stream>>>(qraw, QrB, pos, H);
    rope_kernel<<<dim3(S, KVH, BB), 128, 0, stream>>>(kraw, KrB, pos, KVH);

    // 4) Flash attention (64 q-rows per block, block-shared async K/V staging)
    attn_kernel<<<dim3(S / 64, H, BB), 128, 0, stream>>>(QrB, KrB, VtB, attnB);

    // 5) Output projection -> fp32 d_out (M=4096, N=2304, K=2048)
    gemm_bf16_kernel<2><<<dim3(HID / 128, (int)TOK / 64), 256, 0, stream>>>(
        attnB, Wob, (float*)d_out, (int)TOK, HID, HD);
}
